// transformer_estimator_29240137351717
// MI455X (gfx1250) — compile-verified, matrix operand order checked by reading, not run on hardware
//
#include <hip/hip_runtime.h>
#include <stdint.h>

// ---------------------------------------------------------------------------
// Types for WMMA
// ---------------------------------------------------------------------------
typedef __attribute__((ext_vector_type(16))) _Float16 v16h;
typedef __attribute__((ext_vector_type(8)))  float    v8f;

struct alignas(16) H8 { _Float16 h[8]; };

#define B_SZ  4096
#define N_SZ  512
#define H_SZ  8
#define DH_SZ 64
#define FC1_SZ 512
#define FC2_SZ 256

__device__ __forceinline__ v8f wmma_f16(v16h a, v16h b, v8f c) {
    return __builtin_amdgcn_wmma_f32_16x16x32_f16(
        /*neg_a=*/false, a, /*neg_b=*/false, b,
        /*c_mod=*/(short)0, c, /*reuse_a=*/false, /*reuse_b=*/false);
}

// CDNA5 async global->LDS copy (ASYNCcnt). lds_off = low 32 bits of generic
// LDS address == byte offset in the workgroup LDS allocation (ISA 10.2).
__device__ __forceinline__ void async_ld_b128(unsigned lds_off, const void* g) {
    asm volatile("global_load_async_to_lds_b128 %0, %1, off"
                 :: "v"(lds_off), "v"((unsigned long long)(uintptr_t)g)
                 : "memory");
}
__device__ __forceinline__ void wait_async0() {
    asm volatile("s_wait_asynccnt 0" ::: "memory");
}
__device__ __forceinline__ unsigned lds_off_of(const void* p) {
    return (unsigned)(uintptr_t)p;
}

// A-style fragment load of a 16x32 f16 tile (row-major, row stride = ld halves).
// ISA layout: lane r (r<16) holds row r: K 0..7 (hi=0) / 8..15 (hi=1) in halves
// 0..7, and K 16..23 / 24..31 in halves 8..15.
__device__ __forceinline__ v16h load_fragA(const _Float16* p, int ld, int r, int hi) {
    const _Float16* row = p + r * ld + hi * 8;
    H8 a = *(const H8*)(row);
    H8 b = *(const H8*)(row + 16);
    v16h f;
#pragma unroll
    for (int i = 0; i < 8; ++i) { f[i] = a.h[i]; f[8 + i] = b.h[i]; }
    return f;
}

// ---------------------------------------------------------------------------
// 1) Multi-hot pooling -> item_emb (B x 512) f16
// ---------------------------------------------------------------------------
__device__ __forceinline__ void pool_region(const int* xb, int base, int len,
                                            const float* W, int t, int* flags,
                                            float& acc, float& cnt) {
    for (int c0 = 0; c0 < len; c0 += 128) {
        __syncthreads();
        int j = c0 + t;
        flags[t] = (j < len) ? xb[base + j] : 0;
        __syncthreads();
        int lim = len - c0; if (lim > 128) lim = 128;
        for (int jj = 0; jj < lim; ++jj) {
            if (flags[jj]) {
                acc += W[(size_t)(c0 + jj) * 128 + t];
                cnt += 1.0f;
            }
        }
    }
}

__global__ __launch_bounds__(128) void pool_kernel(
    const int* __restrict__ x, const float* __restrict__ emb_rate,
    const float* __restrict__ Wg, const float* __restrict__ Wd,
    const float* __restrict__ Wa, _Float16* __restrict__ item_emb) {
    __shared__ int flags[128];
    const int b = blockIdx.x;
    const int t = threadIdx.x;
    const int* xb = x + (size_t)b * 10246;

    int rate = xb[0];
    float ag = 0.f, cg = 0.f, ad = 0.f, cd = 0.f, aa = 0.f, ca = 0.f;
    pool_region(xb, 1,    25,   Wg, t, flags, ag, cg);
    pool_region(xb, 26,   2186, Wd, t, flags, ad, cd);
    pool_region(xb, 2212, 8030, Wa, t, flags, aa, ca);

    _Float16* out = item_emb + (size_t)b * N_SZ;
    out[t]       = (_Float16)emb_rate[(size_t)rate * 128 + t];
    out[128 + t] = (_Float16)(ag / cg);
    out[256 + t] = (_Float16)(ad / cd);
    out[384 + t] = (_Float16)(aa / ca);
}

// ---------------------------------------------------------------------------
// 2) user_emb (512) f32 from 4 lookups using x[0, 10242..10245]
// ---------------------------------------------------------------------------
__global__ void user_emb_kernel(const int* __restrict__ x,
                                const float* __restrict__ eg,
                                const float* __restrict__ ea,
                                const float* __restrict__ eo,
                                const float* __restrict__ er,
                                float* __restrict__ user_emb) {
    int t = threadIdx.x;          // 0..511
    int seg = t >> 7, d = t & 127;
    float v;
    if (seg == 0)      v = eg[(size_t)x[10242] * 128 + d];
    else if (seg == 1) v = ea[(size_t)x[10243] * 128 + d];
    else if (seg == 2) v = eo[(size_t)x[10244] * 128 + d];
    else               v = er[(size_t)x[10245] * 128 + d];
    user_emb[t] = v;
}

// ---------------------------------------------------------------------------
// 3) user projections: qu (scaled 1/8) and vu, each (H,64) f32
// ---------------------------------------------------------------------------
__global__ void uproj_kernel(const float* __restrict__ user_emb,
                             const float* __restrict__ Wq_u,
                             const float* __restrict__ Wv_u,
                             float* __restrict__ qu, float* __restrict__ vu) {
    int t = threadIdx.x;          // 0..511 = h*64 + d
    int h = t >> 6, d = t & 63;
    float aq = 0.f, av = 0.f;
    for (int n = 0; n < N_SZ; ++n) {
        float u = user_emb[n];
        size_t w = ((size_t)h * N_SZ + n) * DH_SZ + d;
        aq += u * Wq_u[w];
        av += u * Wv_u[w];
    }
    qu[t] = aq * 0.125f;
    vu[t] = av;
}

// ---------------------------------------------------------------------------
// 4) Batched WMMA GEMM: C[z] = relu(scale * A @ B[z] + bias), f16 out
//    A: (M,K) f16 (shared across z), B: (K,N) f32, C: (M,N) f16
//    block = 256 threads (8 waves), tile 128(M) x 64(N), K-chunks of 32
//    A tile staged via CDNA5 async global->LDS; B^T converted through VGPRs.
// ---------------------------------------------------------------------------
__global__ __launch_bounds__(256) void gemm_wmma_kernel(
    const _Float16* __restrict__ A, const float* __restrict__ Bg,
    _Float16* __restrict__ Cg, const float* __restrict__ bias,
    int M, int N, int K, long long sB, long long sC, float scale, int do_relu) {
    __shared__ _Float16 As[128 * 32];
    __shared__ _Float16 Bts[64 * 32];   // B^T chunk: [n][k]

    const int z = blockIdx.z;
    const float* Bm = Bg + (size_t)z * sB;
    _Float16* Cm = Cg + (size_t)z * sC;

    const int m0 = blockIdx.x * 128;
    const int n0 = blockIdx.y * 64;
    const int tid = threadIdx.x;
    const int w = tid >> 5;
    const int lane = tid & 31;
    const int r = lane & 15, hi = lane >> 4;

    const int ar = tid >> 1, ac = (tid & 1) * 16;
    const unsigned as_off = lds_off_of(As + ar * 32 + ac);
    const int bn = tid & 63, bk = (tid >> 6) * 8;

    v8f acc[4];
#pragma unroll
    for (int n = 0; n < 4; ++n)
#pragma unroll
        for (int v = 0; v < 8; ++v) acc[n][v] = 0.f;

    for (int kc = 0; kc < K; kc += 32) {
        // stage A tile (128 x 32) asynchronously into LDS
        const _Float16* asrc = A + (size_t)(m0 + ar) * K + kc + ac;
        async_ld_b128(as_off, asrc);
        async_ld_b128(as_off + 16, asrc + 8);
        // prefetch next chunk while this one is in flight
        if (kc + 32 < K) {
            __builtin_prefetch(asrc + 32, 0, 1);
            __builtin_prefetch(Bm + (size_t)(kc + 32 + bk) * N + n0 + bn, 0, 1);
        }
        // stage B^T tile (64 n x 32 k), converting f32 -> f16 (overlaps async)
#pragma unroll
        for (int j = 0; j < 8; ++j)
            Bts[bn * 32 + bk + j] =
                (_Float16)Bm[(size_t)(kc + bk + j) * N + n0 + bn];
        wait_async0();
        __syncthreads();

        v16h aA = load_fragA(As + (w * 16) * 32, 32, r, hi);
#pragma unroll
        for (int nt = 0; nt < 4; ++nt) {
            v16h bB = load_fragA(Bts + (nt * 16) * 32, 32, r, hi);
            acc[nt] = wmma_f16(aA, bB, acc[nt]);
        }
        __syncthreads();
    }

    // epilogue: C-layout element (row = v + 8*hi, col = lane&15)
#pragma unroll
    for (int nt = 0; nt < 4; ++nt) {
        int col = n0 + nt * 16 + r;
        float bv = bias ? bias[col] : 0.f;
#pragma unroll
        for (int v = 0; v < 8; ++v) {
            int row = m0 + w * 16 + v + 8 * hi;
            float val = acc[nt][v] * scale + bv;
            if (do_relu) val = val > 0.f ? val : 0.f;
            Cm[(size_t)row * N + col] = (_Float16)val;
        }
    }
}

// ---------------------------------------------------------------------------
// 5) Flash attention per head: v_item = softmax(qi ki^T) vi + vu
//    grid = (B/128, H), block = 256 (8 waves x 16 query rows), 64-key chunks
//    writes directly into X[:, 512 + h*64 + d]
// ---------------------------------------------------------------------------
__global__ __launch_bounds__(256) void flash_attn_kernel(
    const _Float16* __restrict__ qi, const _Float16* __restrict__ ki,
    const _Float16* __restrict__ vi, const float* __restrict__ vu,
    _Float16* __restrict__ X) {
    extern __shared__ _Float16 lds[];
    _Float16* Ks = lds;                        // 64 keys x 64 dims (8 KB)
    _Float16* Vt = lds + 64 * 64;              // 64 dims x 64 keys, V^T (8 KB)
    const int tid = threadIdx.x;
    const int w = tid >> 5;                    // 8 waves
    const int lane = tid & 31;
    const int r = lane & 15, hi = lane >> 4;
    _Float16* Ps = lds + 2 * 64 * 64 + w * (16 * 64);  // per-wave P (2 KB x 8)

    const int h = blockIdx.y;
    const int q0 = blockIdx.x * 128;

    const _Float16* qh = qi + ((size_t)h * B_SZ + q0 + w * 16) * DH_SZ;
    const _Float16* kh = ki + (size_t)h * B_SZ * DH_SZ;
    const _Float16* vh = vi + (size_t)h * B_SZ * DH_SZ;

    // Q fragments (16 rows x 64 dims), qi already pre-scaled by 1/8
    v16h qa0 = load_fragA(qh, DH_SZ, r, hi);
    v16h qa1 = load_fragA(qh + 32, DH_SZ, r, hi);

    v8f o[4];
    float mi[8], li[8];
#pragma unroll
    for (int n = 0; n < 4; ++n)
#pragma unroll
        for (int v = 0; v < 8; ++v) o[n][v] = 0.f;
#pragma unroll
    for (int v = 0; v < 8; ++v) { mi[v] = -1e30f; li[v] = 0.f; }

    const unsigned ks_off = lds_off_of(Ks);
    const int vd = tid & 63, vkb = (tid >> 6) * 16;

    for (int kc = 0; kc < B_SZ; kc += 64) {
        // async-stage K chunk (64x64 f16 = 8 KB = 256 threads x 2 b128)
        async_ld_b128(ks_off + tid * 16,
                      kh + (size_t)kc * DH_SZ + tid * 8);
        async_ld_b128(ks_off + 4096 + tid * 16,
                      kh + (size_t)(kc + 32) * DH_SZ + tid * 8);
        // prefetch next chunk of K and V into L2
        if (kc + 64 < B_SZ) {
            __builtin_prefetch(kh + (size_t)(kc + 64) * DH_SZ + tid * 16, 0, 1);
            __builtin_prefetch(vh + (size_t)(kc + 64) * DH_SZ + tid * 16, 0, 1);
        }
        // stage V^T chunk through VGPRs (transpose), overlapping the async copy
#pragma unroll
        for (int j = 0; j < 16; ++j)
            Vt[vd * 64 + vkb + j] = vh[(size_t)(kc + vkb + j) * DH_SZ + vd];
        wait_async0();
        __syncthreads();

        // S = Q K^T for four 16-key groups
        v8f s[4];
#pragma unroll
        for (int g = 0; g < 4; ++g) {
#pragma unroll
            for (int v = 0; v < 8; ++v) s[g][v] = 0.f;
            v16h kb0 = load_fragA(Ks + (g * 16) * 64, 64, r, hi);
            v16h kb1 = load_fragA(Ks + (g * 16) * 64 + 32, 64, r, hi);
            s[g] = wmma_f16(qa0, kb0, s[g]);
            s[g] = wmma_f16(qa1, kb1, s[g]);
        }

        // online softmax: row reductions across the 16-lane half groups
#pragma unroll
        for (int v = 0; v < 8; ++v) {
            float mv = fmaxf(fmaxf(s[0][v], s[1][v]), fmaxf(s[2][v], s[3][v]));
#pragma unroll
            for (int msk = 8; msk >= 1; msk >>= 1)
                mv = fmaxf(mv, __shfl_xor(mv, msk, 32));
            mv = fmaxf(mv, mi[v]);
            float alpha = __expf(mi[v] - mv);
            mi[v] = mv;
            float rs = 0.f;
#pragma unroll
            for (int g = 0; g < 4; ++g) {
                float p = __expf(s[g][v] - mv);
                s[g][v] = p;
                rs += p;
            }
#pragma unroll
            for (int msk = 8; msk >= 1; msk >>= 1)
                rs += __shfl_xor(rs, msk, 32);
            li[v] = li[v] * alpha + rs;
#pragma unroll
            for (int n = 0; n < 4; ++n) o[n][v] *= alpha;
            // C-layout P -> A-layout via per-wave LDS scratch
#pragma unroll
            for (int g = 0; g < 4; ++g)
                Ps[(v + 8 * hi) * 64 + g * 16 + r] = (_Float16)s[g][v];
        }
        asm volatile("s_wait_dscnt 0" ::: "memory");  // same-wave LDS RAW

        v16h pa0 = load_fragA(Ps, 64, r, hi);
        v16h pa1 = load_fragA(Ps + 32, 64, r, hi);
#pragma unroll
        for (int nt = 0; nt < 4; ++nt) {
            v16h vb0 = load_fragA(Vt + (nt * 16) * 64, 64, r, hi);
            v16h vb1 = load_fragA(Vt + (nt * 16) * 64 + 32, 64, r, hi);
            o[nt] = wmma_f16(pa0, vb0, o[nt]);
            o[nt] = wmma_f16(pa1, vb1, o[nt]);
        }
        __syncthreads();
    }

    // epilogue: out = o/l + vu[h], write into X right half
#pragma unroll
    for (int nt = 0; nt < 4; ++nt) {
        int dim = nt * 16 + r;
        float vub = vu[h * DH_SZ + dim];
#pragma unroll
        for (int v = 0; v < 8; ++v) {
            int row = q0 + w * 16 + v + 8 * hi;
            float val = o[nt][v] / li[v] + vub;
            X[(size_t)row * (2 * N_SZ) + N_SZ + h * DH_SZ + dim] = (_Float16)val;
        }
    }
}

// ---------------------------------------------------------------------------
// 6) user attention: user_att[h*64+d] = vu + sum_k softmax(qu.ki)_k vi[k,d]
//    grid = H, block = 256
// ---------------------------------------------------------------------------
__global__ __launch_bounds__(256) void user_attn_kernel(
    const _Float16* __restrict__ ki, const _Float16* __restrict__ vi,
    const float* __restrict__ qu, const float* __restrict__ vu,
    _Float16* __restrict__ user_att) {
    __shared__ float sbuf[B_SZ];
    __shared__ float red[256];
    __shared__ float qsh[DH_SZ];
    const int h = blockIdx.x;
    const int t = threadIdx.x;
    const _Float16* kh = ki + (size_t)h * B_SZ * DH_SZ;
    const _Float16* vh = vi + (size_t)h * B_SZ * DH_SZ;

    if (t < DH_SZ) qsh[t] = qu[h * DH_SZ + t];
    __syncthreads();

    float lmax = -1e30f;
    for (int k = t; k < B_SZ; k += 256) {
        float s = 0.f;
        const _Float16* kr = kh + (size_t)k * DH_SZ;
        for (int d = 0; d < DH_SZ; ++d) s += qsh[d] * (float)kr[d];
        sbuf[k] = s;
        lmax = fmaxf(lmax, s);
    }
    red[t] = lmax; __syncthreads();
    for (int s = 128; s > 0; s >>= 1) {
        if (t < s) red[t] = fmaxf(red[t], red[t + s]);
        __syncthreads();
    }
    float m = red[0]; __syncthreads();

    float lsum = 0.f;
    for (int k = t; k < B_SZ; k += 256) {
        float e = __expf(sbuf[k] - m);
        sbuf[k] = e;
        lsum += e;
    }
    red[t] = lsum; __syncthreads();
    for (int s = 128; s > 0; s >>= 1) {
        if (t < s) red[t] += red[t + s];
        __syncthreads();
    }
    float l = red[0]; __syncthreads();

    if (t < DH_SZ) {
        float acc = 0.f;
        for (int k = 0; k < B_SZ; ++k)
            acc += sbuf[k] * (float)vh[(size_t)k * DH_SZ + t];
        user_att[h * DH_SZ + t] = (_Float16)(vu[h * DH_SZ + t] + acc / l);
    }
}

// broadcast user_att into X[:, 0:512]
__global__ void bcast_kernel(const _Float16* __restrict__ user_att,
                             _Float16* __restrict__ X) {
    size_t idx = (size_t)blockIdx.x * blockDim.x + threadIdx.x;
    size_t b = idx >> 9;
    int c = (int)(idx & 511);
    if (b < B_SZ) X[b * (2 * N_SZ) + c] = user_att[c];
}

// ---------------------------------------------------------------------------
// 7) final: out[b] = h2[b,:] . W_out + b_out  (f32)
// ---------------------------------------------------------------------------
__global__ __launch_bounds__(256) void final_kernel(
    const _Float16* __restrict__ h2, const float* __restrict__ W_out,
    const float* __restrict__ b_out, float* __restrict__ out) {
    __shared__ float red[256];
    const int b = blockIdx.x, t = threadIdx.x;
    red[t] = (float)h2[(size_t)b * FC2_SZ + t] * W_out[t];
    __syncthreads();
    for (int s = 128; s > 0; s >>= 1) {
        if (t < s) red[t] += red[t + s];
        __syncthreads();
    }
    if (t == 0) out[b] = red[0] + b_out[0];
}

// ---------------------------------------------------------------------------
// Launch
// ---------------------------------------------------------------------------
extern "C" void kernel_launch(void* const* d_in, const int* in_sizes, int n_in,
                              void* d_out, int out_size, void* d_ws, size_t ws_size,
                              hipStream_t stream) {
    const int*   x          = (const int*)d_in[0];
    const float* emb_rate   = (const float*)d_in[1];
    const float* W_genre    = (const float*)d_in[2];
    const float* W_director = (const float*)d_in[3];
    const float* W_actor    = (const float*)d_in[4];
    const float* emb_gender = (const float*)d_in[5];
    const float* emb_age    = (const float*)d_in[6];
    const float* emb_occ    = (const float*)d_in[7];
    const float* emb_area   = (const float*)d_in[8];
    const float* Wq_u       = (const float*)d_in[9];
    /* Wk_u (d_in[10]) is mathematically unused: single-key softmax == 1 */
    const float* Wv_u       = (const float*)d_in[11];
    const float* Wq_i       = (const float*)d_in[12];
    const float* Wk_i       = (const float*)d_in[13];
    const float* Wv_i       = (const float*)d_in[14];
    const float* W1         = (const float*)d_in[15];
    const float* b1         = (const float*)d_in[16];
    const float* W2         = (const float*)d_in[17];
    const float* b2         = (const float*)d_in[18];
    const float* W_out      = (const float*)d_in[19];
    const float* b_out      = (const float*)d_in[20];

    char* ws = (char*)d_ws;
    _Float16* item_emb = (_Float16*)ws; ws += (size_t)B_SZ * N_SZ * 2;
    _Float16* X        = (_Float16*)ws; ws += (size_t)B_SZ * 2 * N_SZ * 2;
    _Float16* h1       = (_Float16*)ws; ws += (size_t)B_SZ * FC1_SZ * 2;
    _Float16* h2       = (_Float16*)ws; ws += (size_t)B_SZ * FC2_SZ * 2;
    _Float16* qi       = (_Float16*)ws; ws += (size_t)H_SZ * B_SZ * DH_SZ * 2;
    _Float16* ki       = (_Float16*)ws; ws += (size_t)H_SZ * B_SZ * DH_SZ * 2;
    _Float16* vi       = (_Float16*)ws; ws += (size_t)H_SZ * B_SZ * DH_SZ * 2;
    float*    user_emb = (float*)ws;    ws += N_SZ * 4;
    float*    qu       = (float*)ws;    ws += H_SZ * DH_SZ * 4;
    float*    vu       = (float*)ws;    ws += H_SZ * DH_SZ * 4;
    _Float16* user_att = (_Float16*)ws; ws += N_SZ * 2;

    // 1) pooling -> item_emb
    pool_kernel<<<B_SZ, 128, 0, stream>>>(x, emb_rate, W_genre, W_director,
                                          W_actor, item_emb);
    // 2) user embedding
    user_emb_kernel<<<1, 512, 0, stream>>>(x, emb_gender, emb_age, emb_occ,
                                           emb_area, user_emb);
    // 3) user projections (qu scaled by 1/8)
    uproj_kernel<<<1, 512, 0, stream>>>(user_emb, Wq_u, Wv_u, qu, vu);

    // 4) item projections via batched WMMA GEMM (z = head)
    long long sB = (long long)N_SZ * DH_SZ;      // 512*64 per head
    long long sC = (long long)B_SZ * DH_SZ;      // 4096*64 per head
    dim3 gp(B_SZ / 128, 1, H_SZ);
    gemm_wmma_kernel<<<gp, 256, 0, stream>>>(item_emb, Wq_i, qi, nullptr,
                                             B_SZ, DH_SZ, N_SZ, sB, sC, 0.125f, 0);
    gemm_wmma_kernel<<<gp, 256, 0, stream>>>(item_emb, Wk_i, ki, nullptr,
                                             B_SZ, DH_SZ, N_SZ, sB, sC, 1.0f, 0);
    gemm_wmma_kernel<<<gp, 256, 0, stream>>>(item_emb, Wv_i, vi, nullptr,
                                             B_SZ, DH_SZ, N_SZ, sB, sC, 1.0f, 0);

    // 5) user attention -> user_att, broadcast into X[:, :512]
    user_attn_kernel<<<H_SZ, 256, 0, stream>>>(ki, vi, qu, vu, user_att);
    bcast_kernel<<<(B_SZ * N_SZ) / 256, 256, 0, stream>>>(user_att, X);

    // 6) flash attention -> X[:, 512:]  (8 waves x 16 rows = 128 queries/WG)
    size_t attn_lds = (2 * 64 * 64 + 8 * 16 * 64) * sizeof(_Float16); // 32 KB
    flash_attn_kernel<<<dim3(B_SZ / 128, H_SZ), 256, attn_lds, stream>>>(
        qi, ki, vi, vu, X);

    // 7) MLP via WMMA GEMM
    gemm_wmma_kernel<<<dim3(B_SZ / 128, FC1_SZ / 64, 1), 256, 0, stream>>>(
        X, W1, h1, b1, B_SZ, FC1_SZ, 2 * N_SZ, 0, 0, 1.0f, 1);
    gemm_wmma_kernel<<<dim3(B_SZ / 128, FC2_SZ / 64, 1), 256, 0, stream>>>(
        h1, W2, h2, b2, B_SZ, FC2_SZ, FC1_SZ, 0, 0, 1.0f, 1);

    // 8) output
    final_kernel<<<B_SZ, 256, 0, stream>>>(h2, W_out, b_out, (float*)d_out);
}
